// AGFAttention_65180423684367
// MI455X (gfx1250) — matrix operand order repaired and checked by reading, not compiled, not run on hardware
//
#include <hip/hip_runtime.h>
#include <hip/hip_bf16.h>
#include <stdint.h>

// Problem constants (match reference)
#define BATCH 4
#define SEQ   1024
#define DIM_  1024
#define NH    16
#define DH_   64
#define ORDER_ 3
#define MAXREL_ 64
#define VOCAB_ 129
#define VOCABP 144          // vocab padded to multiple of 16 for WMMA
#define TOPK_ 256
#define SCALE_ 0.125f       // DH^-0.5

typedef _Float16 f16;
typedef __attribute__((ext_vector_type(16))) _Float16 v16h;
typedef __attribute__((ext_vector_type(8)))  _Float16 v8h;
typedef __attribute__((ext_vector_type(8)))  float    v8f;
typedef __attribute__((ext_vector_type(4)))  unsigned int u32x4;
typedef __attribute__((ext_vector_type(8)))  int          i32x8;
typedef __attribute__((ext_vector_type(4)))  int          i32x4;

#define WMMA_F16(A,B,C) __builtin_amdgcn_wmma_f32_16x16x32_f16(false,(A),false,(B),(short)0,(C),false,false)

// Tensor Data Mover path: this toolchain exposes the 6-arg builtin
// (uint32x4 g0, int32x8 g1, int32x4 g2, int32x4 g3, int32x8 ext, i32 cpol).
#if defined(__HIP_DEVICE_COMPILE__) && __has_builtin(__builtin_amdgcn_tensor_load_to_lds) && __has_builtin(__builtin_amdgcn_s_wait_tensorcnt)
#define USE_TDM 1
#else
#define USE_TDM 0
#endif

// ---------------------------------------------------------------------------
// Fragment loaders (wave32 WMMA; contiguous k order shared by A and B, not the ISA table order)
// A (16x32 f16): lane m=lane&15, half=lane>>4; own k order (not ISA table): VGPR0-3 K = 16*half+0..7,
// VGPR4-7 hold K = 16*half+8..15  -> two contiguous v8h loads per lane.
static __device__ __forceinline__ v16h load_a_frag(const f16* rowk, int lane) {
  const int half = (lane >> 4) & 1;
  v8h x0 = *(const v8h*)(rowk + 16 * half);
  v8h x1 = *(const v8h*)(rowk + 16 * half + 8);
  v16h a;
  #pragma unroll
  for (int i = 0; i < 8; ++i) { a[i] = x0[i]; a[8 + i] = x1[i]; }
  return a;
}

// B (32x16 f16) where source is row-major (N,K): B[k][n] = src[n0+n][k].
// lane n=lane&15 holds column n, K = 16*(lane>>4)+0..15 -> single v16h load.
static __device__ __forceinline__ v16h load_b_rowK(const f16* base, int ld, int n0, int kb, int lane) {
  const int n = n0 + (lane & 15);
  const int k = kb + ((lane >> 4) & 1) * 16;
  return *(const v16h*)(base + (size_t)n * ld + k);
}

static __device__ __forceinline__ uint32_t order_key(float s) {
  uint32_t u = __float_as_uint(s);
  return (u & 0x80000000u) ? ~u : (u | 0x80000000u);
}

// ---------------------------------------------------------------------------
__global__ __launch_bounds__(256) void AGF_cvt_f16(const float* __restrict__ s, f16* __restrict__ d, int n) {
  int i = blockIdx.x * blockDim.x + threadIdx.x;
  if (i < n) d[i] = (f16)s[i];
}

__global__ __launch_bounds__(256) void AGF_prep_small(const float* __restrict__ relk,
                                                      const float* __restrict__ alphas,
                                                      f16* __restrict__ relkh,
                                                      float* __restrict__ alpha_sig) {
  int i = blockIdx.x * blockDim.x + threadIdx.x;
  if (i < VOCABP * DH_) {
    int r = i / DH_;
    relkh[i] = (f16)(r < VOCAB_ ? relk[i] : 0.f);
  }
  if (i < ORDER_ * NH) {
    alpha_sig[i] = 1.f / (1.f + __expf(-alphas[i]));
  }
}

// ---------------------------------------------------------------------------
// QKV GEMM. q,k stored (B,H,N,DH) f16; v stored TRANSPOSED (B,H,DH,N) f16 so
// the apply kernel's B-fragments are single contiguous v16h loads.
__global__ __launch_bounds__(256) void AGF_gemm_qkv(
    const f16* __restrict__ xh, const f16* __restrict__ wh,
    const float* __restrict__ bqkv,
    f16* __restrict__ qh, f16* __restrict__ kh, f16* __restrict__ vTh) {
  const int tid = threadIdx.x, wave = tid >> 5, lane = tid & 31;
  const int m0 = blockIdx.y * 64 + (wave >> 1) * 16;
  const int n0 = blockIdx.x * 128 + (wave & 1) * 64;
  const int m = lane & 15;
  const f16* arow = xh + (size_t)(m0 + m) * DIM_;
  v8f c0 = {}, c1 = {}, c2 = {}, c3 = {};
  for (int kc = 0; kc < DIM_ / 32; ++kc) {
    const int kb = kc * 32;
    v16h a  = load_a_frag(arow + kb, lane);
    v16h b0 = load_b_rowK(wh, DIM_, n0 +  0, kb, lane);
    v16h b1 = load_b_rowK(wh, DIM_, n0 + 16, kb, lane);
    v16h b2 = load_b_rowK(wh, DIM_, n0 + 32, kb, lane);
    v16h b3 = load_b_rowK(wh, DIM_, n0 + 48, kb, lane);
    c0 = WMMA_F16(a, b0, c0); c1 = WMMA_F16(a, b1, c1);
    c2 = WMMA_F16(a, b2, c2); c3 = WMMA_F16(a, b3, c3);
  }
  const int grp = (lane >> 4) & 1, cn = lane & 15;
  #pragma unroll
  for (int r = 0; r < 8; ++r) {
    const int mm = m0 + grp * 8 + r;
    const int bb = mm >> 10, tok = mm & 1023;
    #pragma unroll
    for (int t = 0; t < 4; ++t) {
      v8f& c = t == 0 ? c0 : t == 1 ? c1 : t == 2 ? c2 : c3;
      const int col = n0 + t * 16 + cn;
      const float val = c[r] + bqkv[col];
      const int sec = col >> 10, cc = col & 1023;
      const int hh = cc >> 6, dd = cc & 63;
      if (sec == 0)
        qh[(((size_t)bb * NH + hh) * SEQ + tok) * DH_ + dd] = (f16)val;
      else if (sec == 1)
        kh[(((size_t)bb * NH + hh) * SEQ + tok) * DH_ + dd] = (f16)val;
      else
        vTh[(((size_t)bb * NH + hh) * DH_ + dd) * SEQ + tok] = (f16)val;
    }
  }
}

// ---------------------------------------------------------------------------
// Attention: scores = q@k^T*SCALE + gather(q@relk^T); exact top-k threshold
// (bitwise binary search); masked softmax; attn -> global f16.
// K-tiles staged through LDS by the Tensor Data Mover, double-buffered per
// wave, synchronized with s_wait_tensorcnt.
__global__ __launch_bounds__(256) void AGF_attn(
    const f16* __restrict__ qh, const f16* __restrict__ kh,
    const f16* __restrict__ relkh, f16* __restrict__ attn) {
  __shared__ float s_sc[16][SEQ];      // 64 KB
  __shared__ float s_qrel[16][VOCABP]; // 9 KB

  const int it = blockIdx.x, h = blockIdx.y, b = blockIdx.z;
  const int i0 = it * 16;
  const int tid = threadIdx.x, wave = tid >> 5, lane = tid & 31;
  const size_t headoff = ((size_t)(b * NH + h)) * SEQ * DH_;
  const f16* qhead = qh + headoff;
  const f16* khead = kh + headoff;
  const int m = lane & 15;

  v16h a0 = load_a_frag(qhead + (size_t)(i0 + m) * DH_ + 0,  lane);
  v16h a1 = load_a_frag(qhead + (size_t)(i0 + m) * DH_ + 32, lane);

  const int grp = (lane >> 4) & 1, cn = lane & 15;

  // Phase 1: qrel = q_tile @ rel_k^T  (16 x 144)
  for (int nt = wave; nt < VOCABP / 16; nt += 8) {
    v8f c = {};
    v16h b0 = load_b_rowK(relkh, DH_, nt * 16, 0,  lane);
    v16h b1 = load_b_rowK(relkh, DH_, nt * 16, 32, lane);
    c = WMMA_F16(a0, b0, c);
    c = WMMA_F16(a1, b1, c);
    #pragma unroll
    for (int r = 0; r < 8; ++r) s_qrel[grp * 8 + r][nt * 16 + cn] = c[r];
  }
  __syncthreads();

  // Phase 2: score tiles. Each wave owns 8 of 64 j-tiles.
#if USE_TDM
  __shared__ __align__(32) f16 s_kst[8][2][16 * DH_];  // 32 KB TDM staging
  {
    const uint32_t lds0 = (uint32_t)(uintptr_t)(&s_kst[wave][0][0]);
    const uint32_t lds1 = (uint32_t)(uintptr_t)(&s_kst[wave][1][0]);
    auto tdm_issue = [&](int jt, uint32_t lds_addr) {
      // D# per CDNA5 ISA: 2D tensor 64x1024 f16,
      // tile 64x16 at row jt*16; count=1, type=2 (image), data_size=2B.
      const uint64_t ga = (uint64_t)(uintptr_t)(khead + (size_t)jt * 16 * DH_);
      u32x4 g0 = { 1u, lds_addr, (uint32_t)ga,
                   (uint32_t)((ga >> 32) & 0x1FFFFFFu) | (2u << 30) };
      i32x8 g1 = { (int)(1u << 16),      // data_size = 2B (wg_mask=0)
                   (int)(64u << 16),     // tensor_dim0 = 64  (bits 79:48)
                   (int)(1024u << 16),   // tensor_dim1 = 1024 (bits 111:80)
                   (int)(64u << 16),     // tile_dim0 = 64 (bits 127:112)
                   16,                   // tile_dim1 = 16 (bits 143:128)
                   64,                   // tensor_dim0_stride = 64 (bits 207:160)
                   0, 0 };
      i32x4 gz = { 0, 0, 0, 0 };
      i32x8 gz8 = { 0, 0, 0, 0, 0, 0, 0, 0 };
      __builtin_amdgcn_tensor_load_to_lds(g0, g1, gz, gz, gz8, 0);
    };
    int cur = 0;
    tdm_issue(wave, lds0);
    for (int jt = wave; jt < SEQ / 16; jt += 8) {
      const bool more = (jt + 8) < SEQ / 16;
      if (more) {
        tdm_issue(jt + 8, cur ? lds0 : lds1);
        __builtin_amdgcn_s_wait_tensorcnt(1);   // current tile landed
      } else {
        __builtin_amdgcn_s_wait_tensorcnt(0);
      }
      const f16* kt = &s_kst[wave][cur][0];
      v16h b0 = load_b_rowK(kt, DH_, 0, 0,  lane);
      v16h b1 = load_b_rowK(kt, DH_, 0, 32, lane);
      v8f c = {};
      c = WMMA_F16(a0, b0, c);
      c = WMMA_F16(a1, b1, c);
      const int j0 = jt * 16;
      #pragma unroll
      for (int r = 0; r < 8; ++r) {
        const int mm = grp * 8 + r;
        const int gi = i0 + mm, gj = j0 + cn;
        int d = gi - gj;
        d = d < -MAXREL_ ? -MAXREL_ : (d > MAXREL_ ? MAXREL_ : d);
        s_sc[mm][gj] = c[r] * SCALE_ + s_qrel[mm][d + MAXREL_];
      }
      cur ^= 1;
    }
  }
#else
  for (int jt = wave; jt < SEQ / 16; jt += 8) {
    const int j0 = jt * 16;
    v8f c = {};
    v16h b0 = load_b_rowK(khead, DH_, j0, 0,  lane);
    v16h b1 = load_b_rowK(khead, DH_, j0, 32, lane);
    c = WMMA_F16(a0, b0, c);
    c = WMMA_F16(a1, b1, c);
    #pragma unroll
    for (int r = 0; r < 8; ++r) {
      const int mm = grp * 8 + r;
      const int gi = i0 + mm, gj = j0 + cn;
      int d = gi - gj;
      d = d < -MAXREL_ ? -MAXREL_ : (d > MAXREL_ ? MAXREL_ : d);
      s_sc[mm][gj] = c[r] * SCALE_ + s_qrel[mm][d + MAXREL_];
    }
  }
#endif
  __syncthreads();

  // Phase 3: exact 256th-largest per row (bitwise binary search on order keys)
  const int row = wave * 2 + grp;
  const int l16 = cn;
  uint32_t kth = 0u;
  for (int bit = 31; bit >= 0; --bit) {
    const uint32_t cand = kth | (1u << bit);
    int cnt = 0;
    for (int t = 0; t < SEQ / 16; ++t)
      cnt += (order_key(s_sc[row][l16 + 16 * t]) >= cand) ? 1 : 0;
    for (int o = 8; o >= 1; o >>= 1) cnt += __shfl_xor(cnt, o, 16);
    if (cnt >= TOPK_) kth = cand;
  }

  // Phase 4: masked softmax + write attn (f16)
  float rmax = -3.0e38f;
  for (int t = 0; t < SEQ / 16; ++t) rmax = fmaxf(rmax, s_sc[row][l16 + 16 * t]);
  for (int o = 8; o >= 1; o >>= 1) rmax = fmaxf(rmax, __shfl_xor(rmax, o, 16));
  float sum = 0.f;
  for (int t = 0; t < SEQ / 16; ++t) {
    const int j = l16 + 16 * t;
    const float s = s_sc[row][j];
    const float e = (order_key(s) >= kth) ? __expf(s - rmax) : 0.f;
    s_sc[row][j] = e;
    sum += e;
  }
  for (int o = 8; o >= 1; o >>= 1) sum += __shfl_xor(sum, o, 16);
  const float inv = 1.0f / sum;
  f16* arow = attn + (((size_t)(b * NH + h) * SEQ) + (i0 + row)) * SEQ;
  for (int t = 0; t < SEQ / 16; ++t) {
    const int j = l16 + 16 * t;
    arow[j] = (f16)(s_sc[row][j] * inv);
  }
}

// ---------------------------------------------------------------------------
// Apply: a_v = attn @ v_curr (+ vocab-binned rel_v on ord 0).
// v_curr/v_next in TRANSPOSED (B,H,DH,N) layout -> every B-fragment is one
// contiguous v16h load. Partial C merged across waves via LDS ds_add_f32.
__global__ __launch_bounds__(256) void AGF_apply(
    const f16* __restrict__ attn, const f16* __restrict__ vTcur,
    const float* __restrict__ relv, const float* __restrict__ alpha_sig,
    f16* __restrict__ vTnext, float* __restrict__ res, int ord) {
  __shared__ float s_acc[16][DH_];        // 4 KB
  __shared__ float s_w[16][VOCAB_ + 3];   // 8.25 KB

  const int it = blockIdx.x, h = blockIdx.y, b = blockIdx.z;
  const int i0 = it * 16;
  const int tid = threadIdx.x, wave = tid >> 5, lane = tid & 31;
  const int grp = (lane >> 4) & 1, cn = lane & 15;

  for (int idx = tid; idx < 16 * DH_; idx += 256) ((float*)s_acc)[idx] = 0.f;
  __syncthreads();

  const f16* arow0 = attn + ((size_t)(b * NH + h) * SEQ + i0) * SEQ;
  const f16* vThead = vTcur + ((size_t)(b * NH + h)) * DH_ * SEQ;
  const int m = lane & 15;

  v8f c0 = {}, c1 = {}, c2 = {}, c3 = {};
  for (int kc = wave * 4; kc < wave * 4 + 4; ++kc) {
    const int j0 = kc * 32;
    v16h a  = load_a_frag(arow0 + (size_t)m * SEQ + j0, lane);
    v16h b0 = load_b_rowK(vThead, SEQ,  0, j0, lane);  // B[k][n] = vT[d0+n][j0+k]
    v16h b1 = load_b_rowK(vThead, SEQ, 16, j0, lane);
    v16h b2 = load_b_rowK(vThead, SEQ, 32, j0, lane);
    v16h b3 = load_b_rowK(vThead, SEQ, 48, j0, lane);
    c0 = WMMA_F16(a, b0, c0); c1 = WMMA_F16(a, b1, c1);
    c2 = WMMA_F16(a, b2, c2); c3 = WMMA_F16(a, b3, c3);
  }
  #pragma unroll
  for (int r = 0; r < 8; ++r) {
    const int mm = grp * 8 + r;
    atomicAdd(&s_acc[mm][ 0 + cn], c0[r]);
    atomicAdd(&s_acc[mm][16 + cn], c1[r]);
    atomicAdd(&s_acc[mm][32 + cn], c2[r]);
    atomicAdd(&s_acc[mm][48 + cn], c3[r]);
  }
  __syncthreads();

  if (ord == 0) {
    // Vocab-binned rel_v: wsum[m][bin] = sum of attn over j with dist==bin.
    const int wrow = wave * 2 + grp;
    const int wl = cn;
    const int gi = i0 + wrow;
    const f16* arow = arow0 + (size_t)wrow * SEQ;
    float pre = 0.f, suf = 0.f;
    for (int j = wl; j < SEQ; j += 16) {
      const float av = (float)arow[j];
      if (j <= gi - MAXREL_) pre += av;   // clipped +64 -> bin 128
      if (j >= gi + MAXREL_) suf += av;   // clipped -64 -> bin 0
    }
    for (int o = 8; o >= 1; o >>= 1) { pre += __shfl_xor(pre, o, 16); suf += __shfl_xor(suf, o, 16); }
    if (wl == 0) { s_w[wrow][2 * MAXREL_] = pre; s_w[wrow][0] = suf; }
    for (int bin = 1 + wl; bin < 2 * MAXREL_; bin += 16) {
      const int j = gi - (bin - MAXREL_);
      s_w[wrow][bin] = (j >= 0 && j < SEQ) ? (float)arow[j] : 0.f;
    }
    __syncthreads();
    const int rm = tid >> 4, rl = tid & 15;
    for (int t = 0; t < 4; ++t) {
      const int d = rl + 16 * t;
      float a = 0.f;
      for (int v = 0; v < VOCAB_; ++v) a += s_w[rm][v] * relv[v * DH_ + d];
      s_acc[rm][d] += a;
    }
    __syncthreads();
  }

  // Epilogue: v_next (transposed f16), res accumulate (f32)
  const int em = tid >> 4, el = tid & 15;
  const float alpha = alpha_sig[ord * NH + h];
  const size_t rbase = ((size_t)(b * NH + h) * SEQ + i0 + em) * DH_;
  const size_t vtb   = ((size_t)(b * NH + h)) * DH_ * SEQ;
  for (int t = 0; t < 4; ++t) {
    const int d = el + 16 * t;
    const float a = s_acc[em][d];
    vTnext[vtb + (size_t)d * SEQ + (i0 + em)] = (f16)a;
    res[rbase + d] = (ord == 0 ? 0.f : res[rbase + d]) + alpha * a;
  }
}

// ---------------------------------------------------------------------------
// Output GEMM: out[m][o] = sum_c res(B,H,N,DH)[m,c] * Wout[o][c] + bout[o].
__global__ __launch_bounds__(256) void AGF_gemm_out(
    const float* __restrict__ res, const f16* __restrict__ wouth,
    const float* __restrict__ bout, float* __restrict__ out) {
  const int tid = threadIdx.x, wave = tid >> 5, lane = tid & 31;
  const int m0 = blockIdx.y * 64 + (wave >> 1) * 16;
  const int n0 = blockIdx.x * 128 + (wave & 1) * 64;
  const int m = lane & 15;
  const int mrow = m0 + m;
  const int bb = mrow >> 10, tok = mrow & 1023;
  const int half = (lane >> 4) & 1;
  v8f c0 = {}, c1 = {}, c2 = {}, c3 = {};
  for (int kc = 0; kc < DIM_ / 32; ++kc) {
    const int kb = kc * 32;
    const int hh = kb >> 6, d0 = kb & 63;  // 32-chunk stays inside one head
    const float* p = res + (((size_t)bb * NH + hh) * SEQ + tok) * DH_ + d0;
    v16h a;
    #pragma unroll
    for (int i = 0; i < 8; ++i) { a[i] = (f16)p[16 * half + i]; a[8 + i] = (f16)p[16 * half + 8 + i]; }
    v16h b0 = load_b_rowK(wouth, DIM_, n0 +  0, kb, lane);
    v16h b1 = load_b_rowK(wouth, DIM_, n0 + 16, kb, lane);
    v16h b2 = load_b_rowK(wouth, DIM_, n0 + 32, kb, lane);
    v16h b3 = load_b_rowK(wouth, DIM_, n0 + 48, kb, lane);
    c0 = WMMA_F16(a, b0, c0); c1 = WMMA_F16(a, b1, c1);
    c2 = WMMA_F16(a, b2, c2); c3 = WMMA_F16(a, b3, c3);
  }
  const int grp = (lane >> 4) & 1, cn = lane & 15;
  #pragma unroll
  for (int r = 0; r < 8; ++r) {
    const int mm = m0 + grp * 8 + r;
    #pragma unroll
    for (int t = 0; t < 4; ++t) {
      v8f& c = t == 0 ? c0 : t == 1 ? c1 : t == 2 ? c2 : c3;
      const int col = n0 + t * 16 + cn;
      out[(size_t)mm * DIM_ + col] = c[r] + bout[col];
    }
  }
}

// ---------------------------------------------------------------------------
extern "C" void kernel_launch(void* const* d_in, const int* in_sizes, int n_in,
                              void* d_out, int out_size, void* d_ws, size_t ws_size,
                              hipStream_t stream) {
  const float* x      = (const float*)d_in[0];
  const float* Wqkv   = (const float*)d_in[1];
  const float* bqkv   = (const float*)d_in[2];
  const float* Wout   = (const float*)d_in[3];
  const float* bout   = (const float*)d_in[4];
  const float* relk   = (const float*)d_in[5];
  const float* relv   = (const float*)d_in[6];
  const float* alphas = (const float*)d_in[7];
  float* out = (float*)d_out;
  (void)in_sizes; (void)n_in; (void)out_size; (void)ws_size;

  char* ws = (char*)d_ws;
  size_t off = 0;
  auto alloc = [&](size_t bytes) -> char* {
    char* p = ws + off;
    off = (off + bytes + 255) & ~(size_t)255;
    return p;
  };
  const size_t NTOK = (size_t)BATCH * SEQ;            // 4096
  f16*   xh      = (f16*)  alloc(NTOK * DIM_ * 2);
  f16*   wqkvh   = (f16*)  alloc((size_t)3 * DIM_ * DIM_ * 2);
  f16*   wouth   = (f16*)  alloc((size_t)DIM_ * DIM_ * 2);
  f16*   relkh   = (f16*)  alloc((size_t)VOCABP * DH_ * 2);
  float* alphsig = (float*)alloc((size_t)ORDER_ * NH * 4);
  f16*   qh      = (f16*)  alloc(NTOK * DIM_ * 2);
  f16*   kh      = (f16*)  alloc(NTOK * DIM_ * 2);
  f16*   vTh     = (f16*)  alloc(NTOK * DIM_ * 2);
  f16*   attn    = (f16*)  alloc((size_t)BATCH * NH * SEQ * SEQ * 2);
  f16*   va      = (f16*)  alloc(NTOK * DIM_ * 2);
  f16*   vb      = (f16*)  alloc(NTOK * DIM_ * 2);
  float* res     = (float*)alloc(NTOK * DIM_ * 4);

  {
    int n1 = (int)(NTOK * DIM_);
    AGF_cvt_f16<<<(n1 + 255) / 256, 256, 0, stream>>>(x, xh, n1);
    int n2 = 3 * DIM_ * DIM_;
    AGF_cvt_f16<<<(n2 + 255) / 256, 256, 0, stream>>>(Wqkv, wqkvh, n2);
    int n3 = DIM_ * DIM_;
    AGF_cvt_f16<<<(n3 + 255) / 256, 256, 0, stream>>>(Wout, wouth, n3);
    AGF_prep_small<<<(VOCABP * DH_ + 255) / 256, 256, 0, stream>>>(relk, alphas, relkh, alphsig);
  }

  // QKV projection -> q,k (B,H,N,DH) + v transposed (B,H,DH,N)
  AGF_gemm_qkv<<<dim3(3 * DIM_ / 128, NTOK / 64), 256, 0, stream>>>(xh, wqkvh, bqkv, qh, kh, vTh);

  // Scores + rel_k + exact top-k + softmax -> attn (f16)
  AGF_attn<<<dim3(SEQ / 16, NH, BATCH), 256, 0, stream>>>(qh, kh, relkh, attn);

  // ORDER chained applications (global dependency between iterations)
  AGF_apply<<<dim3(SEQ / 16, NH, BATCH), 256, 0, stream>>>(attn, vTh, relv, alphsig, va, res, 0);
  AGF_apply<<<dim3(SEQ / 16, NH, BATCH), 256, 0, stream>>>(attn, va,  relv, alphsig, vb, res, 1);
  AGF_apply<<<dim3(SEQ / 16, NH, BATCH), 256, 0, stream>>>(attn, vb,  relv, alphsig, va, res, 2);

  // Output projection
  AGF_gemm_out<<<dim3(DIM_ / 128, NTOK / 64), 256, 0, stream>>>(res, wouth, bout, out);
}
